// GraphEmbedding_76390288327606
// MI455X (gfx1250) — 8-bit matrix operand order repaired and checked by reading, compiled, not run on hardware
//
#include <hip/hip_runtime.h>
#include <hip/hip_bf16.h>
#include <math.h>

typedef __attribute__((ext_vector_type(2))) float v2f;
typedef __attribute__((ext_vector_type(8))) float v8f;
typedef __attribute__((ext_vector_type(8))) int   v8i;

#define NN 2048
#define BB 16
#define TT 256

__global__ void k_avg(const float* __restrict__ x, float* __restrict__ avg) {
    int n = blockIdx.x * blockDim.x + threadIdx.x;
    int t = blockIdx.y;
    float s = 0.f;
#pragma unroll
    for (int b = 0; b < BB; ++b)
        s += x[(size_t)(b * TT + t) * NN + n];
    avg[(size_t)n * TT + t] = s * (1.0f / BB);
}

__global__ void k_norm(const float* __restrict__ avg, float* __restrict__ xn) {
    __shared__ float red[256];
    int n = blockIdx.x;
    int t = threadIdx.x;
    float v = avg[(size_t)n * TT + t];
    red[t] = v * v;
    __syncthreads();
    for (int off = 128; off > 0; off >>= 1) {
        if (t < off) red[t] += red[t + off];
        __syncthreads();
    }
    float inv = 1.0f / fmaxf(sqrtf(red[0]), 1e-12f);
    xn[(size_t)n * TT + t] = v * inv;
}

__global__ void k_simadj(const float* __restrict__ xn,
                         const float2* __restrict__ gum,
                         unsigned char* __restrict__ adj) {
    constexpr int LS = TT + 4;
    __shared__ float ldsB[16 * LS];
    int tid  = threadIdx.x;
    int lane = tid & 31;
    int wave = tid >> 5;
    int j0 = blockIdx.x * 16;
    int i0 = (blockIdx.y * 8 + wave) * 16;
    int m = lane & 15;
    int h = lane >> 4;
    {
        int r = tid >> 4;
        int g = (tid & 15) * 16;
        const float* src = xn + (size_t)(j0 + r) * TT + g;
        float* dst = &ldsB[r * LS + g];
#pragma unroll
        for (int q = 0; q < 16; q += 4)
            *(float4*)(dst + q) = *(const float4*)(src + q);
    }
    __syncthreads();
    const float* arow = xn + (size_t)(i0 + m) * TT;
    v8f acc = {};
#pragma unroll 8
    for (int k0 = 0; k0 < TT; k0 += 4) {
        int kk = k0 + 2 * h;
        v2f a = *(const v2f*)(arow + kk);
        v2f b = *(const v2f*)(&ldsB[m * LS + kk]);
        acc = __builtin_amdgcn_wmma_f32_16x16x4_f32(false, a, false, b,
                                                    (short)0, acc, false, false);
    }
    int j = j0 + m;
#pragma unroll
    for (int v = 0; v < 8; ++v) {
        int i = i0 + v + 8 * h;
        float sim = (acc[v] + 1.0f) * 0.5f;
        float2 g = gum[(size_t)i * NN + j];
        float s0 = sim + g.x;
        float s1 = (1.0f - sim) + g.y;
        adj[(size_t)i * NN + j] = (s0 > s1) ? (unsigned char)1 : (unsigned char)0;
    }
}

__global__ void k_nmask(const unsigned char* __restrict__ adj,
                        unsigned char* __restrict__ nm) {
    size_t idx = (size_t)blockIdx.x * blockDim.x + threadIdx.x;
    int i = (int)(idx / NN), j = (int)(idx % NN);
    unsigned char v = (i == j) ? (unsigned char)1
                               : (unsigned char)(adj[idx] | adj[(size_t)j * NN + i]);
    nm[idx] = v;
}

__global__ void k_common(const unsigned char* __restrict__ nm,
                         float* __restrict__ cf, int* __restrict__ maxc) {
    int lane = threadIdx.x & 31;
    int wave = threadIdx.x >> 5;
    int j0 = blockIdx.x * 16;
    int i0 = (blockIdx.y * 8 + wave) * 16;
    int m = lane & 15;
    int h = lane >> 4;
    const unsigned char* arow = nm + (size_t)(i0 + m) * NN;
    const unsigned char* brow = nm + (size_t)(j0 + m) * NN;
    v8i acc = {};
    for (int k0 = 0; k0 < NN; k0 += 64) {
        v8i a, b;
#pragma unroll
        for (int d = 0; d < 8; ++d) {
            int ka = k0 + 4 * (d & 1) + 16 * ((d >> 1) & 1) + 32 * (d >> 2) + 8 * h;
            int kb = k0 + 4 * (d & 1) + 16 * ((d >> 1) & 1) + 32 * (d >> 2) + 8 * h;
            a[d] = *(const int*)(arow + ka);
            b[d] = *(const int*)(brow + kb);
        }
        acc = __builtin_amdgcn_wmma_i32_16x16x64_iu8(false, a, false, b, acc,
                                                     false, false);
    }
    int j = j0 + m;
    int lmax = 0;
#pragma unroll
    for (int v = 0; v < 8; ++v) {
        int i = i0 + v + 8 * h;
        cf[(size_t)i * NN + j] = (float)acc[v];
        lmax = max(lmax, acc[v]);
    }
    for (int off = 16; off > 0; off >>= 1)
        lmax = max(lmax, __shfl_xor(lmax, off));
    if (lane == 0) atomicMax(maxc, lmax);
}

__global__ void k_init(int* maxc) { if (threadIdx.x == 0) *maxc = 0; }

__global__ void k_struct(const unsigned char* __restrict__ adj,
                         float* __restrict__ cf, const int* __restrict__ maxc) {
    size_t idx = (size_t)blockIdx.x * blockDim.x + threadIdx.x;
    int s = (int)(idx / NN), d = (int)(idx % NN);
    float c = cf[idx];
    unsigned char a_sd = adj[idx];
    unsigned char asym = (s == d) ? a_sd : (unsigned char)(a_sd | adj[(size_t)d * NN + s]);
    float mc = (float)(*maxc);
    float st = (asym && (c > 1.0f)) ? (c / mc) * c : 0.0f;
    cf[idx] = a_sd ? st : 0.0f;
}

__global__ void k_deg(const float* __restrict__ W, float* __restrict__ dinv) {
    int d = blockIdx.x * blockDim.x + threadIdx.x;
    float s = 0.f;
    for (int ss = 0; ss < NN; ++ss) s += W[(size_t)ss * NN + d];
    dinv[d] = (s > 0.0f) ? rsqrtf(s) : 0.0f;
}

__global__ void k_wnt(const float* __restrict__ W, const float* __restrict__ dinv,
                      float* __restrict__ WnT) {
    size_t idx = (size_t)blockIdx.x * blockDim.x + threadIdx.x;
    int d = (int)(idx / NN), s = (int)(idx % NN);
    WnT[idx] = dinv[s] * W[(size_t)s * NN + d] * dinv[d];
}

__global__ void k_xw(const float* __restrict__ x, const float* __restrict__ wgt,
                     float* __restrict__ xwT) {
    constexpr int LS = TT + 4;
    __shared__ float ldsW[16 * LS];
    int tid  = threadIdx.x;
    int lane = tid & 31;
    int wave = tid >> 5;
    int t0 = blockIdx.x * 16;
    int n0 = (blockIdx.y * 8 + wave) * 16;
    int b  = blockIdx.z;
    int m = lane & 15;
    int h = lane >> 4;
    {
        int n = tid & 15;
        int kbase = (tid >> 4) * 16;
#pragma unroll
        for (int q = 0; q < 16; ++q)
            ldsW[n * LS + kbase + q] = wgt[(size_t)(kbase + q) * TT + t0 + n];
    }
    __syncthreads();
    v8f acc = {};
    for (int k0 = 0; k0 < TT; k0 += 4) {
        int kb = k0 + 2 * h;
        v2f a, w;
        a[0] = x[(size_t)(b * TT + kb)     * NN + n0 + m];
        a[1] = x[(size_t)(b * TT + kb + 1) * NN + n0 + m];
        w = *(const v2f*)(&ldsW[m * LS + kb]);
        acc = __builtin_amdgcn_wmma_f32_16x16x4_f32(false, a, false, w,
                                                    (short)0, acc, false, false);
    }
#pragma unroll
    for (int v = 0; v < 8; ++v)
        xwT[(size_t)(b * TT + t0 + m) * NN + n0 + v + 8 * h] = acc[v];
}

__global__ void k_agg(const float* __restrict__ WnT, const float* __restrict__ xwT,
                      const float* __restrict__ bias, float* __restrict__ out) {
    constexpr int KC = 256;
    constexpr int LS = KC + 4;
    __shared__ float ldsB[16 * LS];
    int tid  = threadIdx.x;
    int lane = tid & 31;
    int wave = tid >> 5;
    int c0 = blockIdx.x * 16;
    int d0 = (blockIdx.y * 8 + wave) * 16;
    int m = lane & 15;
    int h = lane >> 4;
    const float* arow = WnT + (size_t)(d0 + m) * NN;
    int r = tid >> 4;
    int g = (tid & 15) * 16;
    const float* bsrc = xwT + (size_t)(c0 + r) * NN + g;
    float* bdst = &ldsB[r * LS + g];
    v8f acc = {};
    for (int k0 = 0; k0 < NN; k0 += KC) {
        __syncthreads();
#pragma unroll
        for (int q = 0; q < 16; q += 4)
            *(float4*)(bdst + q) = *(const float4*)(bsrc + k0 + q);
        __syncthreads();
        if (k0 + KC < NN)
            __builtin_prefetch(arow + k0 + KC, 0, 1);
#pragma unroll 8
        for (int kk = 0; kk < KC; kk += 4) {
            int kq = kk + 2 * h;
            v2f a = *(const v2f*)(arow + k0 + kq);
            v2f b = *(const v2f*)(&ldsB[m * LS + kq]);
            acc = __builtin_amdgcn_wmma_f32_16x16x4_f32(false, a, false, b,
                                                        (short)0, acc, false, false);
        }
    }
    int c = c0 + m;
    float bv = bias[c & (TT - 1)];
#pragma unroll
    for (int v = 0; v < 8; ++v)
        out[(size_t)c * NN + d0 + v + 8 * h] = acc[v] + bv;
}

extern "C" void kernel_launch(void* const* d_in, const int* in_sizes, int n_in,
                              void* d_out, int out_size, void* d_ws, size_t ws_size,
                              hipStream_t stream) {
    const float*  x    = (const float*)d_in[0];
    const float*  wgt  = (const float*)d_in[1];
    const float*  bias = (const float*)d_in[2];
    const float2* gum  = (const float2*)d_in[3];

    char* ws = (char*)d_ws;
    float*         avg = (float*)(ws + (size_t)0);
    float*         xn  = (float*)(ws + ((size_t)2  << 20));
    unsigned char* adj = (unsigned char*)(ws + ((size_t)4  << 20));
    unsigned char* nm  = (unsigned char*)(ws + ((size_t)8  << 20));
    float*         cf  = (float*)(ws + ((size_t)12 << 20));
    float*         xwT = (float*)(ws + (size_t)0);
    float*         WnT = (float*)(ws + ((size_t)32 << 20));
    float*         dinv= (float*)(ws + ((size_t)48 << 20));
    int*           maxc= (int*)  (ws + ((size_t)48 << 20) + 8192);

    k_init  <<<1, 64, 0, stream>>>(maxc);
    k_avg   <<<dim3(NN / 256, TT), 256, 0, stream>>>(x, avg);
    k_norm  <<<NN, TT, 0, stream>>>(avg, xn);
    k_simadj<<<dim3(NN / 16, NN / 128), 256, 0, stream>>>(xn, gum, adj);
    k_nmask <<<(NN * NN) / 256, 256, 0, stream>>>(adj, nm);
    k_common<<<dim3(NN / 16, NN / 128), 256, 0, stream>>>(nm, cf, maxc);
    k_struct<<<(NN * NN) / 256, 256, 0, stream>>>(adj, cf, maxc);
    k_deg   <<<NN / 256, 256, 0, stream>>>(cf, dinv);
    k_wnt   <<<(NN * NN) / 256, 256, 0, stream>>>(cf, dinv, WnT);
    k_xw    <<<dim3(TT / 16, NN / 128, BB), 256, 0, stream>>>(x, wgt, xwT);
    k_agg   <<<dim3((BB * TT) / 16, NN / 128), 256, 0, stream>>>(WnT, xwT, bias, (float*)d_out);
}
